// SingleHeadCrossAttention_20624432955539
// MI455X (gfx1250) — matrix operand order repaired and checked by reading, not compiled, not run on hardware
//
#include <hip/hip_runtime.h>
#include <hip/hip_bf16.h>

// Problem constants (match reference)
#define BQ   256      // batch of queries
#define DD   128      // feature dim
#define D1   32       // param feature dim
#define HID  64
#define TOPK 16
#define CHUNK 512     // keys per chunk in the score pass
#define GWG  256      // persistent workgroups for the score pass

typedef __attribute__((ext_vector_type(16))) __bf16 v16bf;
typedef __attribute__((ext_vector_type(8)))  float  v8f;

// ---------------------------------------------------------------------------
// Kernel 1a: Wqk = Wq @ Wk^T   (128x128, trivial)
// ---------------------------------------------------------------------------
__global__ void k_wqk(const float* __restrict__ Wq, const float* __restrict__ Wk,
                      float* __restrict__ Wqk) {
  int idx = blockIdx.x * 256 + threadIdx.x;       // 16384 outputs
  int i = idx >> 7, j = idx & 127;
  float s = 0.f;
  #pragma unroll 8
  for (int k = 0; k < DD; ++k) s += Wq[i * DD + k] * Wk[j * DD + k];
  Wqk[i * DD + j] = s;
}

// ---------------------------------------------------------------------------
// Kernel 1b: Qp = query @ Wqk (fp32 + bf16 copy for WMMA A operand)
// ---------------------------------------------------------------------------
__global__ void k_qp(const float* __restrict__ query, const float* __restrict__ Wqk,
                     float* __restrict__ Qp, __bf16* __restrict__ Qpb) {
  int idx = blockIdx.x * 256 + threadIdx.x;       // 32768 outputs
  int b = idx >> 7, d = idx & 127;
  float s = 0.f;
  #pragma unroll 8
  for (int k = 0; k < DD; ++k) s += query[b * DD + k] * Wqk[k * DD + d];
  Qp[idx]  = s;
  Qpb[idx] = (__bf16)s;
}

// ---------------------------------------------------------------------------
// Kernel 2: streaming scores via v_wmma_f32_16x16x32_bf16 + per-lane top-16.
// 256 threads = 8 waves. Wave w owns queries [32w, 32w+32) as two 16-row
// WMMA M-tiles. Grid-strides over key chunks; each lane keeps a running
// top-16 (score,index) for one query in registers.
// ---------------------------------------------------------------------------
__global__ __launch_bounds__(256) void k_scores(
    const __bf16* __restrict__ Qpb, const float* __restrict__ keys,
    float* __restrict__ cand_s, int* __restrict__ cand_i, int N) {
  __shared__ float sbuf[8][32][16];               // per-wave score bounce (16KB)

  const int tid  = threadIdx.x;
  const int wave = tid >> 5;
  const int lane = tid & 31;
  const int l    = lane & 15;
  const int hi   = lane >> 4;

  // ---- A fragments: 2 query blocks x 4 K-slices, held in VGPRs for the
  // whole kernel. ISA 16-bit A layout: lane<16 -> M=lane, K={0..7}u{16..23}.
  v16bf afrag[2][4];
  #pragma unroll
  for (int blk = 0; blk < 2; ++blk) {
    const int row = (wave * 2 + blk) * 16 + l;
    #pragma unroll
    for (int ksi = 0; ksi < 4; ++ksi) {
      const __bf16* ap = Qpb + row * DD + ksi * 32 + hi * 8;
      v16bf a;
      #pragma unroll
      for (int e = 0; e < 8; ++e) { a[e] = ap[e]; a[8 + e] = ap[16 + e]; }
      afrag[blk][ksi] = a;
    }
  }

  // Running top-16 in registers (all indices constant after unrolling).
  float tv[TOPK]; int ti[TOPK];
  #pragma unroll
  for (int i = 0; i < TOPK; ++i) { tv[i] = -3.0e38f; ti[i] = -1; }
  float runMin = -3.0e38f;

  const int nchunks = (N + CHUNK - 1) / CHUNK;
  for (int ch = blockIdx.x; ch < nchunks; ch += gridDim.x) {
    const int kbase = ch * CHUNK;
    for (int kt = 0; kt < CHUNK; kt += 16) {
      const int krow  = kbase + kt + l;
      const bool valid = krow < N;

      // Prefetch a future key row (global_prefetch_b8); speculative, guarded.
      if (kbase + kt + 128 + l < N)
        __builtin_prefetch(keys + (size_t)(kbase + kt + 128 + l) * DD, 0, 0);

      // ---- B fragments: lane<16 -> N(key)=lane, K={0..7}u{16..23};
      // lane>=16 -> same key row, K={8..15}u{24..31}. fp32 -> bf16 convert in regs.
      v16bf bfrag[4];
      #pragma unroll
      for (int ksi = 0; ksi < 4; ++ksi) {
        v16bf bv;
        if (valid) {
          const float* kp = keys + (size_t)krow * DD + ksi * 32 + hi * 8;
          #pragma unroll
          for (int e = 0; e < 8; ++e) { bv[e] = (__bf16)kp[e]; bv[8 + e] = (__bf16)kp[16 + e]; }
        } else {
          #pragma unroll
          for (int e = 0; e < 16; ++e) bv[e] = (__bf16)0.0f;
        }
        bfrag[ksi] = bv;
      }

      // ---- 8 WMMAs per 16-key tile (2 blocks x 4 K-slices)
      #pragma unroll
      for (int blk = 0; blk < 2; ++blk) {
        v8f c;
        #pragma unroll
        for (int r = 0; r < 8; ++r) c[r] = 0.0f;
        #pragma unroll
        for (int ksi = 0; ksi < 4; ++ksi)
          c = __builtin_amdgcn_wmma_f32_16x16x32_bf16(
                false, afrag[blk][ksi], false, bfrag[ksi],
                (short)0, c, false, false);
        // C/D layout: lane<16 -> N=lane, M=r ; lane>=16 -> N=lane-16, M=r+8
        #pragma unroll
        for (int r = 0; r < 8; ++r)
          sbuf[wave][blk * 16 + hi * 8 + r][l] = c[r];
      }

      // Intra-wave LDS visibility (same-wave DS ops are in-order; force the
      // compiler ordering + drain DScnt before cross-lane reads).
      __builtin_amdgcn_wave_barrier();
      asm volatile("s_wait_dscnt 0" ::: "memory");

      // ---- read back this lane's 16 scores (4 x b128 LDS loads)
      const float4* srow = (const float4*)(&sbuf[wave][lane][0]);
      float4 r0 = srow[0], r1 = srow[1], r2 = srow[2], r3 = srow[3];
      float sc[16];
      sc[0]=r0.x;  sc[1]=r0.y;  sc[2]=r0.z;  sc[3]=r0.w;
      sc[4]=r1.x;  sc[5]=r1.y;  sc[6]=r1.z;  sc[7]=r1.w;
      sc[8]=r2.x;  sc[9]=r2.y;  sc[10]=r2.z; sc[11]=r2.w;
      sc[12]=r3.x; sc[13]=r3.y; sc[14]=r3.z; sc[15]=r3.w;

      // ---- tile-max early-out: skip merge unless something can enter top-16
      float tmax = sc[0];
      #pragma unroll
      for (int n = 1; n < 16; ++n) tmax = fmaxf(tmax, sc[n]);

      if (tmax > runMin) {
        #pragma unroll
        for (int n = 0; n < 16; ++n) {
          const float s  = sc[n];
          const int  key = kbase + kt + n;
          if (key < N && s > runMin) {
            // branchless replace-the-min (cndmask chains, no nested branches)
            bool done = false;
            #pragma unroll
            for (int i = 0; i < TOPK; ++i) {
              const bool eq = (!done) & (tv[i] == runMin);
              tv[i] = eq ? s   : tv[i];
              ti[i] = eq ? key : ti[i];
              done  = done | eq;
            }
            runMin = tv[0];
            #pragma unroll
            for (int i = 1; i < TOPK; ++i) runMin = fminf(runMin, tv[i]);
          }
        }
      }
      __builtin_amdgcn_wave_barrier();
    }
  }

  const int q = wave * 32 + lane;                 // this lane's query id
  #pragma unroll
  for (int i = 0; i < TOPK; ++i) {
    cand_s[((size_t)blockIdx.x * BQ + q) * TOPK + i] = tv[i];
    cand_i[((size_t)blockIdx.x * BQ + q) * TOPK + i] = ti[i];
  }
}

// ---------------------------------------------------------------------------
// Kernel 3: per-query merge of GWG candidate lists -> global top-16 indices
// ---------------------------------------------------------------------------
__global__ __launch_bounds__(256) void k_merge(
    const float* __restrict__ cand_s, const int* __restrict__ cand_i,
    int* __restrict__ top_i) {
  const int q = blockIdx.x;
  const int t = threadIdx.x;
  const int M = GWG * TOPK;                       // 4096 candidates
  __shared__ float s[GWG * TOPK];
  __shared__ int   id[GWG * TOPK];
  __shared__ float rv[256];
  __shared__ int   rp[256];

  for (int j = t; j < M; j += 256) {
    int wg = j >> 4, i = j & 15;
    float v = cand_s[((size_t)wg * BQ + q) * TOPK + i];
    int  ix = cand_i[((size_t)wg * BQ + q) * TOPK + i];
    s[j]  = (ix < 0) ? -3.0e38f : v;
    id[j] = ix;
  }
  __syncthreads();

  for (int round = 0; round < TOPK; ++round) {
    float bv = -3.0e38f; int bp = -1;
    for (int j = t; j < M; j += 256)
      if (s[j] > bv) { bv = s[j]; bp = j; }
    rv[t] = bv; rp[t] = bp;
    __syncthreads();
    for (int off = 128; off > 0; off >>= 1) {
      if (t < off && rv[t + off] > rv[t]) { rv[t] = rv[t + off]; rp[t] = rp[t + off]; }
      __syncthreads();
    }
    if (t == 0) {
      int p = rp[0];
      top_i[q * TOPK + round] = (p >= 0) ? id[p] : 0;
      if (p >= 0) s[p] = -3.0e38f;
    }
    __syncthreads();
  }
}

// ---------------------------------------------------------------------------
// Kernel 4: gather top-16, V_top = values[idx]@Wv, exact fp32 logits,
// adapter MLP + layernorm + softmax-weighted sum.  One block per query.
// ---------------------------------------------------------------------------
__global__ __launch_bounds__(128) void k_final(
    const float* __restrict__ Qp, const float* __restrict__ keys,
    const float* __restrict__ values, const float* __restrict__ pf,
    const float* __restrict__ Wv, const float* __restrict__ A1w,
    const float* __restrict__ A1b, const float* __restrict__ A2w,
    const float* __restrict__ A2b, const float* __restrict__ lng,
    const float* __restrict__ lnb, const int* __restrict__ top_i,
    float* __restrict__ out, int N) {
  const int q = blockIdx.x;
  const int t = threadIdx.x;                      // 128 threads, t == output dim
  __shared__ int   idx[TOPK];
  __shared__ float vrows[TOPK][DD];
  __shared__ float vt[TOPK][DD];
  __shared__ float logit[TOPK];
  __shared__ float h[HID];
  __shared__ float o[DD];

  if (t < TOPK) idx[t] = top_i[q * TOPK + t];
  __syncthreads();

  for (int e = t; e < TOPK * DD; e += 128) {
    int j = e >> 7, d = e & 127;
    vrows[j][d] = values[(size_t)idx[j] * DD + d];
  }
  __syncthreads();

  // V_top column t
  {
    float acc[TOPK];
    #pragma unroll
    for (int j = 0; j < TOPK; ++j) acc[j] = 0.f;
    for (int k = 0; k < DD; ++k) {
      float w = Wv[k * DD + t];
      #pragma unroll
      for (int j = 0; j < TOPK; ++j) acc[j] += vrows[j][k] * w;
    }
    #pragma unroll
    for (int j = 0; j < TOPK; ++j) vt[j][t] = acc[j];
  }

  // Exact fp32 logits: Qp[q] . keys[idx_j]   (== Q . K_top mathematically)
  if (t < TOPK) {
    const float* kr = keys + (size_t)idx[t] * DD;
    const float* qp = Qp + q * DD;
    float s = 0.f;
    #pragma unroll 8
    for (int k = 0; k < DD; ++k) s += qp[k] * kr[k];
    logit[t] = s * (1.0f / sqrtf((float)DD));
  }
  __syncthreads();

  // softmax weights (computed redundantly per thread; tiny)
  float m = logit[0];
  #pragma unroll
  for (int j = 1; j < TOPK; ++j) m = fmaxf(m, logit[j]);
  float wj[TOPK]; float den = 0.f;
  #pragma unroll
  for (int j = 0; j < TOPK; ++j) { wj[j] = __expf(logit[j] - m); den += wj[j]; }
  const float inv = 1.0f / den;

  float outacc = 0.f;
  for (int j = 0; j < TOPK; ++j) {
    if (t < HID) {
      float a = A1b[t];
      for (int i = 0; i < DD; ++i) a += vt[j][i] * A1w[i * HID + t];
      for (int i = 0; i < D1; ++i) a += pf[q * D1 + i] * A1w[(DD + i) * HID + t];
      h[t] = fmaxf(a, 0.f);
    }
    __syncthreads();
    float od = A2b[t];
    #pragma unroll 8
    for (int i = 0; i < HID; ++i) od += h[i] * A2w[i * DD + t];
    o[t] = od;
    __syncthreads();
    float mu = 0.f;
    for (int i = 0; i < DD; ++i) mu += o[i];
    mu *= (1.0f / DD);
    float var = 0.f;
    for (int i = 0; i < DD; ++i) { float d = o[i] - mu; var += d * d; }
    var *= (1.0f / DD);
    float ln = (od - mu) * rsqrtf(var + 1e-5f) * lng[t] + lnb[t];
    outacc += (wj[j] * inv) * (vt[j][t] + ln);
    __syncthreads();                              // h/o reused next j
  }
  out[q * DD + t] = outacc;
}

// ---------------------------------------------------------------------------
// Launch.  Workspace layout (bytes):
//   [0,      65536)  Wqk      (128x128 f32)
//   [65536,  196608) Qp       (256x128 f32)
//   [196608, 262144) Qp bf16  (256x128)
//   [262144, +4MB)   cand_s   (GWG x 256 x 16 f32)
//   [.. ,    +4MB)   cand_i   (GWG x 256 x 16 i32)
//   [.. ,    +16KB)  top_i    (256 x 16 i32)      total ~8.7 MB
// ---------------------------------------------------------------------------
extern "C" void kernel_launch(void* const* d_in, const int* in_sizes, int n_in,
                              void* d_out, int out_size, void* d_ws, size_t ws_size,
                              hipStream_t stream) {
  const float* query  = (const float*)d_in[0];
  const float* keys   = (const float*)d_in[1];
  const float* values = (const float*)d_in[2];
  const float* pf     = (const float*)d_in[3];
  const float* Wq     = (const float*)d_in[4];
  const float* Wk     = (const float*)d_in[5];
  const float* Wv     = (const float*)d_in[6];
  const float* A1w    = (const float*)d_in[7];
  const float* A1b    = (const float*)d_in[8];
  const float* A2w    = (const float*)d_in[9];
  const float* A2b    = (const float*)d_in[10];
  const float* lng    = (const float*)d_in[11];
  const float* lnb    = (const float*)d_in[12];
  const int N = in_sizes[1] / DD;

  char* ws = (char*)d_ws;
  float*  Wqk    = (float*) (ws + 0);
  float*  Qp     = (float*) (ws + 65536);
  __bf16* Qpb    = (__bf16*)(ws + 196608);
  float*  cand_s = (float*) (ws + 262144);
  int*    cand_i = (int*)   (ws + 262144 + (size_t)GWG * BQ * TOPK * 4);
  int*    top_i  = (int*)   (ws + 262144 + (size_t)GWG * BQ * TOPK * 8);
  float*  outp   = (float*)d_out;

  k_wqk   <<<64,  256, 0, stream>>>(Wq, Wk, Wqk);
  k_qp    <<<128, 256, 0, stream>>>(query, Wqk, Qp, Qpb);
  k_scores<<<GWG, 256, 0, stream>>>(Qpb, keys, cand_s, cand_i, N);
  k_merge <<<BQ,  256, 0, stream>>>(cand_s, cand_i, top_i);
  k_final <<<BQ,  128, 0, stream>>>(Qp, keys, values, pf, Wv, A1w, A1b,
                                    A2w, A2b, lng, lnb, top_i, outp, N);
}
